// SSA_65841848648272
// MI455X (gfx1250) — matrix operand order repaired and checked by reading, compiled, not run on hardware
//
#include <hip/hip_runtime.h>
#include <hip/hip_bf16.h>

typedef __attribute__((ext_vector_type(16))) _Float16 v16h;
typedef __attribute__((ext_vector_type(8)))  _Float16 v8h;
typedef __attribute__((ext_vector_type(8)))  float    v8f;

// ---------------------------------------------------------------------------
// WMMA wrapper: D = A(16x32 f16) * B(32x16 f16) + C(16x16 f32)
// ---------------------------------------------------------------------------
__device__ __forceinline__ v8f wmma_f16(v16h a, v16h b, v8f c) {
  return __builtin_amdgcn_wmma_f32_16x16x32_f16(
      /*neg_a=*/false, a, /*neg_b=*/false, b,
      /*c_mod=*/(short)0, c, /*reuse_a=*/false, /*reuse_b=*/false);
}

__device__ __forceinline__ v16h combine8(v8h lo, v8h hi) {
  v16h r;
#pragma unroll
  for (int i = 0; i < 8; ++i) { r[i] = lo[i]; r[i + 8] = hi[i]; }
  return r;
}

// A-matrix 16x32 (MxK), contiguous-k layout used here (same k order as B):
//   lane l: M = l&15 ; halves i=0..15 -> K = (l>>4)*16 + i
__device__ __forceinline__ v16h load_a_tile(const _Float16* base, int ld) {
  int l = threadIdx.x & 31;
  const _Float16* p = base + (size_t)(l & 15) * ld + ((l >> 4) << 4);
  v8h lo = *reinterpret_cast<const v8h*>(p);
  v8h hi = *reinterpret_cast<const v8h*>(p + 8);
  return combine8(lo, hi);
}

// B-matrix 32x16 (KxN), contiguous-k layout used here (same k order as A):
//   lane l: N = l&15 ; halves i=0..15 -> K = (l>>4)*16 + i
// src is row-major [N][K] (each src row is one B column); base at (n0,k0).
__device__ __forceinline__ v16h load_b_tile(const _Float16* base, int ld) {
  int l = threadIdx.x & 31;
  const _Float16* p = base + (size_t)(l & 15) * ld + ((l >> 4) << 4);
  v8h lo = *reinterpret_cast<const v8h*>(p);
  v8h hi = *reinterpret_cast<const v8h*>(p + 8);
  return combine8(lo, hi);
}

// ---------------------------------------------------------------------------
// fp32 -> fp16 convert
// ---------------------------------------------------------------------------
__global__ void ssa_cvt_f16(const float* __restrict__ in, _Float16* __restrict__ out, int n) {
  int i = blockIdx.x * blockDim.x + threadIdx.x;
  if (i < n) out[i] = (_Float16)in[i];
}

// ---------------------------------------------------------------------------
// GEMM (M x 512) * (512 x 512)^T with fused BatchNorm epilogue.
// Weight strip (128 cols x 32 k) staged block-cooperatively into LDS with
// global_load_async_to_lds_b128 (ASYNCcnt), double buffered so the copy of
// chunk k+32 overlaps the WMMAs of chunk k. A tiles register-pipelined.
// mode 0: store f32 row-major (pre-LIF activations)
// mode 1: store f16 transposed into Vt[tbh][d][n]
// mode 2: store f32 row-major into final output
// Block: 256 thr = 8 waves; wave (wid>>1, wid&1) -> 16x64 strip of 64x128 tile.
// ---------------------------------------------------------------------------
#define BROW 40  // padded LDS row (halves): 128 rows x 40 -> conflict-free b128 reads

__global__ void ssa_gemm_bn(const _Float16* __restrict__ A, const _Float16* __restrict__ W,
                            const float* __restrict__ bias, const float* __restrict__ gam,
                            const float* __restrict__ bet, const float* __restrict__ mu,
                            const float* __restrict__ var,
                            float* __restrict__ outf, _Float16* __restrict__ outh, int mode) {
  __shared__ __align__(16) _Float16 bstage[2][128 * BROW];  // 2 x 10 KB
  const int K = 512;
  int tid = threadIdx.x;
  int lane = tid & 31, wid = tid >> 5;
  int m0  = blockIdx.x * 64 + (wid >> 1) * 16;
  int nblk = blockIdx.y * 128;
  int n0l  = (wid & 1) * 64;  // this wave's column base inside the 128 strip

  // Block-cooperative async copy: 128 cols x 32 halves (8 KB) = 512 x 16B chunks.
  auto stage_b = [&](int buf, int k0) {
#pragma unroll
    for (int j = 0; j < 2; ++j) {
      int q   = tid * 2 + j;        // 0..511
      int c   = q >> 2;             // column 0..127
      int off = (q & 3) * 8;        // half-offset within the 32-half chunk
      const _Float16* g = W + (size_t)(nblk + c) * K + k0 + off;
      unsigned lds = (unsigned)(size_t)&bstage[buf][c * BROW + off];
      asm volatile("global_load_async_to_lds_b128 %0, %1, off"
                   :: "v"(lds), "v"(g) : "memory");
    }
  };

  // B tile out of LDS: two ds_load_b128 per lane, conflict-free (row pad 80B).
  auto load_b_lds = [&](int buf, int c0) {
    const _Float16* p = &bstage[buf][(c0 + (lane & 15)) * BROW + ((lane >> 4) << 4)];
    v8h lo = *reinterpret_cast<const v8h*>(p);
    v8h hi = *reinterpret_cast<const v8h*>(p + 8);
    return combine8(lo, hi);
  };

  stage_b(0, 0);
  v16h a_next = load_a_tile(A + (size_t)m0 * K, K);
  asm volatile("s_wait_asynccnt 0" ::: "memory");
  __syncthreads();

  v8f acc[4] = {};
  for (int k0 = 0; k0 < K; k0 += 32) {
    int buf  = (k0 >> 5) & 1;
    bool more = (k0 + 32) < K;
    if (more) stage_b(buf ^ 1, k0 + 32);        // async: overlaps WMMAs below
    v16h a = a_next;
    if (more) a_next = load_a_tile(A + (size_t)m0 * K + k0 + 32, K);
#pragma unroll
    for (int j = 0; j < 4; ++j) {
      v16h b = load_b_lds(buf, n0l + 16 * j);
      acc[j] = wmma_f16(a, b, acc[j]);
    }
    if (more) {
      asm volatile("s_wait_asynccnt 0" ::: "memory");
      __syncthreads();
    }
  }

  int lrow = lane >> 4;   // D layout: M = r + 8*lrow, N = lane&15
  int lcol = lane & 15;
#pragma unroll
  for (int j = 0; j < 4; ++j) {
    int col = nblk + n0l + 16 * j + lcol;
    float s   = gam[col] * rsqrtf(var[col] + 1e-5f);
    float off = (bias[col] - mu[col]) * s + bet[col];
    if (mode == 1) {
      int m  = m0 + lrow * 8;         // first row of this lane's 8-row group
      int tb = m >> 10, n = m & 1023;
      int h  = col >> 6, d = col & 63;
      union { _Float16 h8[8]; int4 v; } u;
#pragma unroll
      for (int r = 0; r < 8; ++r) u.h8[r] = (_Float16)(acc[j][r] * s + off);
      _Float16* dst = outh + (((size_t)tb * 8 + h) * 64 + d) * 1024 + n;
      *reinterpret_cast<int4*>(dst) = u.v;
    } else {
#pragma unroll
      for (int r = 0; r < 8; ++r) {
        int m = m0 + lrow * 8 + r;
        outf[(size_t)m * 512 + col] = acc[j][r] * s + off;
      }
    }
  }
}

// ---------------------------------------------------------------------------
// Multi-step LIF over T=4: v=(v+u)/2 ; spike=(v>=1) ; hard reset
// ---------------------------------------------------------------------------
__global__ void ssa_lif(const float* __restrict__ u, _Float16* __restrict__ sp, int bnc) {
  int i = blockIdx.x * blockDim.x + threadIdx.x;
  if (i >= bnc) return;
  float v = 0.f;
#pragma unroll
  for (int t = 0; t < 4; ++t) {
    v = 0.5f * (v + u[(size_t)t * bnc + i]);
    float s = (v >= 1.0f) ? 1.0f : 0.0f;
    sp[(size_t)t * bnc + i] = (_Float16)s;
    v *= (1.0f - s);
  }
}

// ---------------------------------------------------------------------------
// Fused attention: O = (Q K^T * 0.125) V per (t,b,h), never materializing S.
// One wave owns 16 Q-rows. S 16x16 tiles go through a per-wave LDS staging
// buffer to convert WMMA D-layout (f32) -> A-layout (f16) for the second GEMM.
// Q,K: [tb][n][512] fp16 (head h at col h*64). Vt: [tbh][d][1024] fp16.
// ---------------------------------------------------------------------------
__global__ void ssa_attn(const _Float16* __restrict__ Q, const _Float16* __restrict__ Kk,
                         const _Float16* __restrict__ Vt, _Float16* __restrict__ O) {
  __shared__ __align__(16) _Float16 stage[8][16 * 32];
  int lane = threadIdx.x & 31, wid = threadIdx.x >> 5;
  int gw  = blockIdx.x * 8 + wid;        // 8192 waves total
  int tbh = gw >> 6;                     // 0..127
  int n0  = (gw & 63) << 4;              // Q-row block
  int tb  = tbh >> 3, h = tbh & 7;
  int lrow = lane >> 4, lcol = lane & 15;

  const _Float16* Qp = Q  + (size_t)tb * 1024 * 512 + h * 64;
  const _Float16* Kp = Kk + (size_t)tb * 1024 * 512 + h * 64;
  const _Float16* Vp = Vt + (size_t)tbh * 64 * 1024;

  v16h qa0 = load_a_tile(Qp + (size_t)n0 * 512,      512);   // d = 0..31
  v16h qa1 = load_a_tile(Qp + (size_t)n0 * 512 + 32, 512);   // d = 32..63
  v8f acc[4] = {};

  for (int mb = 0; mb < 1024; mb += 32) {
#pragma unroll
    for (int sub = 0; sub < 2; ++sub) {
      int m0 = mb + sub * 16;
      v16h kb0 = load_b_tile(Kp + (size_t)m0 * 512,      512);
      v16h kb1 = load_b_tile(Kp + (size_t)m0 * 512 + 32, 512);
      v8f s = {};
      s = wmma_f16(qa0, kb0, s);
      s = wmma_f16(qa1, kb1, s);
      // D-layout -> LDS as [qrow][m_local] f16, scaled by 0.125
#pragma unroll
      for (int r = 0; r < 8; ++r) {
        int qrow = r + 8 * lrow;
        stage[wid][qrow * 32 + sub * 16 + lcol] = (_Float16)(s[r] * 0.125f);
      }
    }
    // re-read the 16x32 S block in A layout (wave-private LDS, DS in-order)
    v16h sa = load_a_tile(&stage[wid][0], 32);
#pragma unroll
    for (int j = 0; j < 4; ++j) {
      v16h vb = load_b_tile(Vp + (size_t)(j * 16) * 1024 + mb, 1024);
      acc[j] = wmma_f16(sa, vb, acc[j]);
    }
  }

  // O tile: M=q rows, N=d; write back into [tb][n][512] fp16 at col h*64+d
#pragma unroll
  for (int j = 0; j < 4; ++j) {
    int d = j * 16 + lcol;
#pragma unroll
    for (int r = 0; r < 8; ++r) {
      int n = n0 + r + 8 * lrow;
      O[((size_t)tb * 1024 + n) * 512 + h * 64 + d] = (_Float16)acc[j][r];
    }
  }
}

// ---------------------------------------------------------------------------
extern "C" void kernel_launch(void* const* d_in, const int* in_sizes, int n_in,
                              void* d_out, int out_size, void* d_ws, size_t ws_size,
                              hipStream_t stream) {
  (void)in_sizes; (void)n_in; (void)out_size; (void)ws_size;
  const float* x    = (const float*)d_in[0];
  // d_in[1] = s_x, unused by the reference forward
  const float* q_w = (const float*)d_in[2];
  const float* q_b = (const float*)d_in[3];
  const float* q_g = (const float*)d_in[4];
  const float* q_bt = (const float*)d_in[5];
  const float* q_m = (const float*)d_in[6];
  const float* q_v = (const float*)d_in[7];
  const float* k_w = (const float*)d_in[8];
  const float* k_b = (const float*)d_in[9];
  const float* k_g = (const float*)d_in[10];
  const float* k_bt = (const float*)d_in[11];
  const float* k_m = (const float*)d_in[12];
  const float* k_v = (const float*)d_in[13];
  const float* v_w = (const float*)d_in[14];
  const float* v_b = (const float*)d_in[15];
  const float* v_g = (const float*)d_in[16];
  const float* v_bt = (const float*)d_in[17];
  const float* v_m = (const float*)d_in[18];
  const float* v_v = (const float*)d_in[19];
  const float* p_w = (const float*)d_in[20];
  const float* p_b = (const float*)d_in[21];
  const float* p_g = (const float*)d_in[22];
  const float* p_bt = (const float*)d_in[23];
  const float* p_m = (const float*)d_in[24];
  const float* p_v = (const float*)d_in[25];

  const size_t M   = 16384;        // T*B*N
  const size_t C   = 512;
  const size_t MC  = M * C;        // 8,388,608
  const size_t WW  = C * C;        // 262,144
  const size_t BNC = 4 * 1024 * 512;

  char* w = (char*)d_ws;
  auto carve = [&](size_t bytes) -> void* {
    void* p = (void*)w;
    w += (bytes + 255) & ~(size_t)255;
    return p;
  };
  _Float16* xh = (_Float16*)carve(MC * 2);
  _Float16* wq = (_Float16*)carve(WW * 2);
  _Float16* wk = (_Float16*)carve(WW * 2);
  _Float16* wv = (_Float16*)carve(WW * 2);
  _Float16* wp = (_Float16*)carve(WW * 2);
  float*    uq = (float*)carve(MC * 4);
  float*    uk = (float*)carve(MC * 4);
  _Float16* Qs = (_Float16*)carve(MC * 2);
  _Float16* Ks = (_Float16*)carve(MC * 2);
  _Float16* Vt = (_Float16*)carve(MC * 2);
  _Float16* Ob = (_Float16*)carve(MC * 2);

  // 1) fp16 conversions
  ssa_cvt_f16<<<(int)(MC / 256), 256, 0, stream>>>(x, xh, (int)MC);
  ssa_cvt_f16<<<(int)(WW / 256), 256, 0, stream>>>(q_w, wq, (int)WW);
  ssa_cvt_f16<<<(int)(WW / 256), 256, 0, stream>>>(k_w, wk, (int)WW);
  ssa_cvt_f16<<<(int)(WW / 256), 256, 0, stream>>>(v_w, wv, (int)WW);
  ssa_cvt_f16<<<(int)(WW / 256), 256, 0, stream>>>(p_w, wp, (int)WW);

  // 2) Q/K/V projections + BN (V written pre-transposed per head)
  dim3 gg((unsigned)(M / 64), 4);
  ssa_gemm_bn<<<gg, 256, 0, stream>>>(xh, wq, q_b, q_g, q_bt, q_m, q_v, uq, nullptr, 0);
  ssa_gemm_bn<<<gg, 256, 0, stream>>>(xh, wk, k_b, k_g, k_bt, k_m, k_v, uk, nullptr, 0);
  ssa_gemm_bn<<<gg, 256, 0, stream>>>(xh, wv, v_b, v_g, v_bt, v_m, v_v, nullptr, Vt, 1);

  // 3) LIF spiking (sequential over T)
  ssa_lif<<<(int)(BNC / 256), 256, 0, stream>>>(uq, Qs, (int)BNC);
  ssa_lif<<<(int)(BNC / 256), 256, 0, stream>>>(uk, Ks, (int)BNC);

  // 4) fused attention (128 tbh batches x 64 row-blocks = 8192 waves)
  ssa_attn<<<1024, 256, 0, stream>>>(Qs, Ks, Vt, Ob);

  // 5) output projection + BN -> f32 d_out
  ssa_gemm_bn<<<gg, 256, 0, stream>>>(Ob, wp, p_b, p_g, p_bt, p_m, p_v,
                                      (float*)d_out, nullptr, 2);
}
